// MultiHeadAttention_910533067476
// MI455X (gfx1250) — matrix operand order repaired and checked by reading, not compiled, not run on hardware
//
#include <hip/hip_runtime.h>
#include <hip/hip_bf16.h>

// ---------------- problem constants ----------------
#define DM   768      // d_model
#define SEQ  2048     // sequence length
#define NB   2        // batch
#define NH   12       // heads
#define HD   64       // head dim
#define MTOT (NB*SEQ) // 4096 rows

typedef __attribute__((ext_vector_type(16))) __bf16 v16bf;
typedef __attribute__((ext_vector_type(8)))  __bf16 v8bf;
typedef __attribute__((ext_vector_type(8)))  float  v8f;

static __device__ __forceinline__ __bf16 f2bf(float f) {
    unsigned u = __builtin_bit_cast(unsigned, f);
    unsigned r = u + 0x7FFFu + ((u >> 16) & 1u);   // RNE
    unsigned short h = (unsigned short)(r >> 16);
    return __builtin_bit_cast(__bf16, h);
}

static __device__ __forceinline__ v16bf mk16(v8bf a, v8bf b) {
    return __builtin_shufflevector(a, b, 0,1,2,3,4,5,6,7,8,9,10,11,12,13,14,15);
}

static __device__ __forceinline__ v8f wmma_bf16(v16bf a, v16bf b, v8f c) {
    return __builtin_amdgcn_wmma_f32_16x16x32_bf16(false, a, false, b, (short)0, c, false, false);
}

// row reductions across the 16 lanes holding one C-tile row (wave32)
static __device__ __forceinline__ float rmax16(float v) {
#pragma unroll
    for (int m = 1; m < 16; m <<= 1) v = fmaxf(v, __shfl_xor(v, m, 32));
    return v;
}
static __device__ __forceinline__ float rsum16(float v) {
#pragma unroll
    for (int m = 1; m < 16; m <<= 1) v += __shfl_xor(v, m, 32);
    return v;
}

// ---------------- fp32 -> bf16 convert ----------------
__global__ __launch_bounds__(256) void k_cvt(const float* __restrict__ s,
                                             __bf16* __restrict__ d, int n) {
    int i = blockIdx.x * blockDim.x + threadIdx.x;
    int stride = gridDim.x * blockDim.x;
    for (; i < n; i += stride) d[i] = f2bf(s[i]);
}

// ---------------- GEMM: Y[m,e] = sum_d X[m,d] * W[e,d] ----------------
// mode 0: Q proj + RoPE  -> dbf as [B,H,S,HD]
// mode 1: K proj + RoPE  -> dbf as [B,H,S,HD]
// mode 2: V proj         -> dbf transposed [B,H,HD,S]
// mode 3: O proj         -> df fp32 [M, DM]
__global__ __launch_bounds__(256)
void k_gemm(const __bf16* __restrict__ X, const __bf16* __restrict__ W,
            const int* __restrict__ pos, __bf16* __restrict__ dbf,
            float* __restrict__ df, int mode) {
    const int lane = threadIdx.x & 31;
    const int wave = threadIdx.x >> 5;
    const int lr   = lane & 15;
    const int half = lane >> 4;
    const int m0 = blockIdx.x * 128 + wave * 16;   // 4096/128 = 32 blocks in x
    const int n0 = blockIdx.y * 64;                // 768/64   = 12 blocks in y

    const v8f z = {0.f,0.f,0.f,0.f,0.f,0.f,0.f,0.f};
    v8f acc[4];
#pragma unroll
    for (int i = 0; i < 4; ++i) acc[i] = z;

    // A fragment rows: lanes 0-15 and 16-31 both cover M=0..15, K-halves differ
    const __bf16* arow = X + (size_t)(m0 + lr) * DM;
    for (int k = 0; k < DM; k += 32) {
        v8bf a0 = *(const v8bf*)(arow + k + half * 16);
        v8bf a1 = *(const v8bf*)(arow + k + half * 16 + 8);
        v16bf A = mk16(a0, a1);
#pragma unroll
        for (int nt = 0; nt < 4; ++nt) {
            const __bf16* bp = W + (size_t)(n0 + nt * 16 + lr) * DM + k + half * 16;
            v16bf Bv = *(const v16bf*)bp;          // 16 contiguous bf16 along d
            acc[nt] = wmma_bf16(A, Bv, acc[nt]);
        }
    }

    // ---- epilogue: C-tile layout is row M = half*8+r, col N = lr ----
#pragma unroll
    for (int nt = 0; nt < 4; ++nt) {
        const int e = n0 + nt * 16 + lr;
        const int h = e >> 6, j = e & 63;
        if (mode <= 1) {
            // RoPE: pair (even,odd) dims sit on adjacent lanes (e, e^1)
            const int p = j >> 1;
            const float invf = __expf(-(float)(2 * p) * (9.210340371976184f / 64.0f));
#pragma unroll
            for (int r = 0; r < 8; ++r) {
                const int m = m0 + half * 8 + r;
                const int b = m >> 11, s = m & (SEQ - 1);
                const float ang = (float)pos[(b << 11) + s] * invf;
                float sn, cs; __sincosf(ang, &sn, &cs);
                float x = acc[nt][r];
                float part = __shfl_xor(x, 1, 32);
                float res = (j & 1) ? (part * sn + x * cs) : (x * cs - part * sn);
                dbf[(((size_t)(b * NH + h)) * SEQ + s) * HD + j] = f2bf(res);
            }
        } else if (mode == 2) {
#pragma unroll
            for (int r = 0; r < 8; ++r) {
                const int m = m0 + half * 8 + r;
                const int b = m >> 11, s = m & (SEQ - 1);
                dbf[(((size_t)(b * NH + h)) * HD + j) * SEQ + s] = f2bf(acc[nt][r]);
            }
        } else {
#pragma unroll
            for (int r = 0; r < 8; ++r) {
                const int m = m0 + half * 8 + r;
                df[(size_t)m * DM + e] = acc[nt][r];
            }
        }
    }
}

// ---------------- flash attention (causal), per-wave 16 q-rows ----------------
__global__ __launch_bounds__(256)
void k_attn(const __bf16* __restrict__ Qh, const __bf16* __restrict__ Kh,
            const __bf16* __restrict__ Vt, __bf16* __restrict__ Ab) {
    __shared__ __align__(32) __bf16 pst[8][16 * 32];   // per-wave P staging, 8 KB
    const int lane = threadIdx.x & 31;
    const int wave = threadIdx.x >> 5;
    const int lr   = lane & 15;
    const int half = lane >> 4;
    const int q0 = blockIdx.x * 128 + wave * 16;
    const int bh = blockIdx.y;

    const __bf16* Qp = Qh + (size_t)bh * SEQ * HD;
    const __bf16* Kp = Kh + (size_t)bh * SEQ * HD;
    const __bf16* Vp = Vt + (size_t)bh * HD * SEQ;

    // Q A-fragments (persist in registers): K-dim = d, two 32-wide steps
    v16bf qa[2];
#pragma unroll
    for (int t = 0; t < 2; ++t) {
        const __bf16* qp = Qp + (size_t)(q0 + lr) * HD + t * 32 + half * 16;
        qa[t] = mk16(*(const v8bf*)qp, *(const v8bf*)(qp + 8));
    }

    const v8f z = {0.f,0.f,0.f,0.f,0.f,0.f,0.f,0.f};
    float mrow[8], lrow[8];
    v8f o[4];
#pragma unroll
    for (int r = 0; r < 8; ++r) { mrow[r] = -INFINITY; lrow[r] = 0.f; }
#pragma unroll
    for (int i = 0; i < 4; ++i) o[i] = z;

    const float sscale = 0.125f;                 // 1/sqrt(64)
    const int kend = q0 + 16;                    // causal: keys <= q0+15
    for (int kb = 0; kb < kend; kb += 32) {
        // ---- S = Q K^T for 32 keys: 2 n-tiles x 2 d-steps ----
        v8f sc[2]; sc[0] = z; sc[1] = z;
#pragma unroll
        for (int t = 0; t < 2; ++t) {
#pragma unroll
            for (int nt = 0; nt < 2; ++nt) {
                const __bf16* kp = Kp + (size_t)(kb + nt * 16 + lr) * HD + t * 32 + half * 16;
                v16bf Bv = *(const v16bf*)kp;
                sc[nt] = wmma_bf16(qa[t], Bv, sc[nt]);
            }
        }
        // ---- online softmax (rows across lanes, cols = keys on lanes) ----
        float pnew[2][8], corr[8];
#pragma unroll
        for (int r = 0; r < 8; ++r) {
            const int q = q0 + half * 8 + r;
            float s0 = sc[0][r] * sscale; if (kb + lr > q)      s0 = -1e30f;
            float s1 = sc[1][r] * sscale; if (kb + 16 + lr > q) s1 = -1e30f;
            float rm = rmax16(fmaxf(s0, s1));
            float mn = fmaxf(mrow[r], rm);
            float p0 = __expf(s0 - mn), p1 = __expf(s1 - mn);
            float rs = rsum16(p0 + p1);
            corr[r] = __expf(mrow[r] - mn);
            lrow[r] = lrow[r] * corr[r] + rs;
            mrow[r] = mn;
            pnew[0][r] = p0; pnew[1][r] = p1;
        }
        // ---- C-layout -> A-layout transpose through LDS ----
#pragma unroll
        for (int nt = 0; nt < 2; ++nt)
#pragma unroll
            for (int r = 0; r < 8; ++r)
                pst[wave][(half * 8 + r) * 32 + nt * 16 + lr] = f2bf(pnew[nt][r]);
        asm volatile("s_wait_dscnt 0x0" ::: "memory");
        const __bf16* pr = &pst[wave][lr * 32 + half * 16];
        v16bf Pa = mk16(*(const v8bf*)pr, *(const v8bf*)(pr + 8));
        // ---- O = O*corr + P V  (V transposed: keys contiguous) ----
#pragma unroll
        for (int nt = 0; nt < 4; ++nt) {
#pragma unroll
            for (int r = 0; r < 8; ++r) o[nt][r] *= corr[r];
            const __bf16* vp = Vp + (size_t)(nt * 16 + lr) * SEQ + kb + half * 16;
            v16bf Bv = *(const v16bf*)vp;
            o[nt] = wmma_bf16(Pa, Bv, o[nt]);
        }
    }

    // ---- normalize and write attention output [B,S,DM] bf16 ----
    const int b = bh / NH, h = bh % NH;
#pragma unroll
    for (int nt = 0; nt < 4; ++nt) {
        const int j = nt * 16 + lr;
#pragma unroll
        for (int r = 0; r < 8; ++r) {
            const int s = q0 + half * 8 + r;
            float x = o[nt][r] / lrow[r];
            Ab[((size_t)(b * SEQ + s)) * DM + h * HD + j] = f2bf(x);
        }
    }
}

// ---------------- host launch ----------------
extern "C" void kernel_launch(void* const* d_in, const int* in_sizes, int n_in,
                              void* d_out, int out_size, void* d_ws, size_t ws_size,
                              hipStream_t stream) {
    const float* x  = (const float*)d_in[0];
    const float* wq = (const float*)d_in[1];
    const float* wk = (const float*)d_in[2];
    const float* wv = (const float*)d_in[3];
    const float* wo = (const float*)d_in[4];
    const int*  pos = (const int*)d_in[5];
    float* out = (float*)d_out;

    const int XN = MTOT * DM;       // 3,145,728
    const int WN = DM * DM;         //   589,824
    const int QN = NB * NH * SEQ * HD; // == XN

    __bf16* base = (__bf16*)d_ws;
    __bf16* Xbf = base;
    __bf16* Wqb = Xbf + XN;
    __bf16* Wkb = Wqb + WN;
    __bf16* Wvb = Wkb + WN;
    __bf16* Wob = Wvb + WN;
    __bf16* Qhp = Wob + WN;
    __bf16* Khp = Qhp + QN;
    __bf16* Vtp = Khp + QN;
    __bf16* Abp = Vtp + QN;

    k_cvt<<<2048, 256, 0, stream>>>(x,  Xbf, XN);
    k_cvt<<<512,  256, 0, stream>>>(wq, Wqb, WN);
    k_cvt<<<512,  256, 0, stream>>>(wk, Wkb, WN);
    k_cvt<<<512,  256, 0, stream>>>(wv, Wvb, WN);
    k_cvt<<<512,  256, 0, stream>>>(wo, Wob, WN);

    dim3 gg(MTOT / 128, DM / 64);   // (32, 12)
    k_gemm<<<gg, 256, 0, stream>>>(Xbf, Wqb, pos, Qhp, nullptr, 0);
    k_gemm<<<gg, 256, 0, stream>>>(Xbf, Wkb, pos, Khp, nullptr, 1);
    k_gemm<<<gg, 256, 0, stream>>>(Xbf, Wvb, pos, Vtp, nullptr, 2);

    dim3 ga(SEQ / 128, NB * NH);    // (16, 24)
    k_attn<<<ga, 256, 0, stream>>>(Qhp, Khp, Vtp, Abp);

    k_gemm<<<gg, 256, 0, stream>>>(Abp, Wob, pos, nullptr, out, 3);
}
